// Transformer_20890720928223
// MI455X (gfx1250) — matrix operand order repaired and checked by reading, not compiled, not run on hardware
//
#include <hip/hip_runtime.h>
#include <hip/hip_bf16.h>
#include <math.h>

typedef _Float16       v16h  __attribute__((ext_vector_type(16)));
typedef float          v8f   __attribute__((ext_vector_type(8)));
typedef unsigned short u16v8 __attribute__((ext_vector_type(8)));

union H16 { v16h v; unsigned short u[16]; u16v8 h[2]; };

static __device__ __forceinline__ unsigned short f2h_bits(float x) {
  union { _Float16 f; unsigned short u; } cv; cv.f = (_Float16)x; return cv.u;
}
static __device__ __forceinline__ float gelu_exact(float x) {
  return 0.5f * x * (1.0f + erff(x * 0.7071067811865475f));
}
static __device__ __forceinline__ v8f vzero() {
  v8f z = {0.f,0.f,0.f,0.f,0.f,0.f,0.f,0.f}; return z;
}

// ---------------- LayerNorm: f32 [row, C] -> f16-bits [row, C] ----------------
__global__ __launch_bounds__(256)
void ln_kernel(const float* __restrict__ x, const float* __restrict__ g,
               const float* __restrict__ b, unsigned short* __restrict__ out, int C)
{
  __shared__ float s1[256], s2[256];
  const int row = blockIdx.x;
  const float* xr = x + (size_t)row * C;
  float sum = 0.f, sq = 0.f;
  for (int i = threadIdx.x; i < C; i += 256) { float v = xr[i]; sum += v; sq += v * v; }
  s1[threadIdx.x] = sum; s2[threadIdx.x] = sq;
  __syncthreads();
  for (int st = 128; st > 0; st >>= 1) {
    if (threadIdx.x < st) { s1[threadIdx.x] += s1[threadIdx.x + st]; s2[threadIdx.x] += s2[threadIdx.x + st]; }
    __syncthreads();
  }
  const float mu  = s1[0] / C;
  const float var = s2[0] / C - mu * mu;
  const float inv = rsqrtf(var + 1e-5f);
  for (int i = threadIdx.x; i < C; i += 256) {
    float v = (xr[i] - mu) * inv * g[i] + b[i];
    out[(size_t)row * C + i] = f2h_bits(v);
  }
}

// ------- Weight transpose+convert: W[K,Nc] f32 -> Wt[Nc,K] f16-bits -------
__global__ __launch_bounds__(256)
void wtrans_kernel(const float* __restrict__ W, unsigned short* __restrict__ Wt,
                   int K, int Nc)
{
  __shared__ unsigned short tile[64][68];
  const int k0 = blockIdx.x * 64, n0 = blockIdx.y * 64;
  #pragma unroll
  for (int i = 0; i < 16; i++) {
    const int flat = threadIdx.x + i * 256;
    const int r = flat >> 6, c = flat & 63;            // r = k-local, c = n-local
    tile[r][c] = f2h_bits(W[(size_t)(k0 + r) * Nc + n0 + c]);
  }
  __syncthreads();
  #pragma unroll
  for (int i = 0; i < 16; i++) {
    const int flat = threadIdx.x + i * 256;
    const int rr = flat >> 6, cc = flat & 63;          // rr = n-local, cc = k-local
    Wt[(size_t)(n0 + rr) * K + k0 + cc] = tile[cc][rr];
  }
}

// ------- V transpose per head: qkv16 V-part [n,d] -> vt[bh][d][n] (f16) -------
__global__ __launch_bounds__(256)
void vtrans_kernel(const unsigned short* __restrict__ qkv, unsigned short* __restrict__ vt)
{
  const int N = 1024, C3 = 3072, D = 64;
  __shared__ unsigned short tile[64][68];
  const int bh = blockIdx.x;            // b*16 + h
  const int n0 = blockIdx.y * 64;
  const int bb = bh >> 4, h = bh & 15;
  const size_t inb = (size_t)bb * N * C3 + 2 * 1024 + (size_t)h * D;
  #pragma unroll
  for (int i = 0; i < 16; i++) {
    const int flat = threadIdx.x + i * 256;
    const int r = flat >> 6, c = flat & 63;            // r = n-local, c = d
    tile[r][c] = qkv[inb + (size_t)(n0 + r) * C3 + c];
  }
  __syncthreads();
  #pragma unroll
  for (int i = 0; i < 16; i++) {
    const int flat = threadIdx.x + i * 256;
    const int rr = flat >> 6, cc = flat & 63;          // rr = d, cc = n-local
    vt[(size_t)bh * D * N + (size_t)rr * N + n0 + cc] = tile[cc][rr];
  }
}

// ---- GEMM: A[M,K] f16-bits x Wt[Nc,K] f16-bits -> out[M,Nc] ----
// Block = 256 thr = 8 waves, block tile 128x128, wave tile 32x64 (2x4 WMMA frags).
template<int ACT_GELU, int OUT_F16>
__global__ __launch_bounds__(256)
void gemm_f16(const unsigned short* __restrict__ A, const unsigned short* __restrict__ Wt,
              const float* __restrict__ bias, const float* __restrict__ resid,
              void* __restrict__ outp, int M, int K, int Nc)
{
  const int lane = threadIdx.x & 31;
  const int wave = threadIdx.x >> 5;
  const int half = lane >> 4;
  const int ln   = lane & 15;
  const int m_base = blockIdx.x * 128 + (wave >> 1) * 32;
  const int n_base = blockIdx.y * 128 + (wave & 1) * 64;

  v8f acc[2][4];
  #pragma unroll
  for (int i = 0; i < 2; i++)
    #pragma unroll
    for (int j = 0; j < 4; j++) acc[i][j] = vzero();

  const unsigned short* aRow0 = A + (size_t)(m_base + ln) * K + 16 * half;
  const unsigned short* aRow1 = A + (size_t)(m_base + 16 + ln) * K + 16 * half;

  for (int k0 = 0; k0 < K; k0 += 32) {
    H16 a[2];
    {
      const u16v8* p0 = (const u16v8*)(aRow0 + k0);
      const u16v8* p1 = (const u16v8*)(aRow1 + k0);
      a[0].h[0] = p0[0]; a[0].h[1] = p0[1];   // k = k0+16*half+e
      a[1].h[0] = p1[0]; a[1].h[1] = p1[1];
    }
    H16 bf[4];
    #pragma unroll
    for (int j = 0; j < 4; j++) {
      // B-frag: n fixed per lane; k = k0 + 16*half + e (contiguous in Wt)
      const u16v8* wp = (const u16v8*)(Wt + (size_t)(n_base + j * 16 + ln) * K + k0 + 16 * half);
      bf[j].h[0] = wp[0];
      bf[j].h[1] = wp[1];
    }
    #pragma unroll
    for (int i = 0; i < 2; i++)
      #pragma unroll
      for (int j = 0; j < 4; j++)
        acc[i][j] = __builtin_amdgcn_wmma_f32_16x16x32_f16(
            false, a[i].v, false, bf[j].v, (short)0, acc[i][j], false, false);
  }

  #pragma unroll
  for (int i = 0; i < 2; i++) {
    #pragma unroll
    for (int j = 0; j < 4; j++) {
      #pragma unroll
      for (int r = 0; r < 8; r++) {
        const int m = m_base + i * 16 + r + 8 * half;
        const int n = n_base + j * 16 + ln;
        float v = acc[i][j][r];
        if (bias)     v += bias[n];
        if (ACT_GELU) v  = gelu_exact(v);
        if (resid)    v += resid[(size_t)m * Nc + n];
        if (OUT_F16) ((unsigned short*)outp)[(size_t)m * Nc + n] = f2h_bits(v);
        else         ((float*)outp)[(size_t)m * Nc + n] = v;
      }
    }
  }
}

// ---------------- Flash attention (no-max softmax; |S| <= ~1 so exp is safe) ----
// qkv[B,N,3C] f16-bits (Q,K used directly), vt[b*h][D][N] f16-bits (V transposed).
// Block = 128 thr = 4 waves; each wave owns 32 query rows (2 16-row frags).
__global__ __launch_bounds__(128)
void attn_kernel(const unsigned short* __restrict__ qkv, const unsigned short* __restrict__ vt,
                 unsigned short* __restrict__ o, float scale)
{
  const int N = 1024, H = 16, D = 64, C = 1024, C3 = 3072;
  const int lane = threadIdx.x & 31;
  const int wave = threadIdx.x >> 5;
  const int half = lane >> 4;
  const int ln   = lane & 15;
  const int NT = N / 128;                 // 8 blocks per (b,h)
  const int bid = blockIdx.x;
  const int nt = bid % NT;
  const int h  = (bid / NT) % H;
  const int bb = bid / (NT * H);
  const int q0 = nt * 128 + wave * 32;
  const size_t base   = (size_t)bb * N * C3 + (size_t)h * D;
  const size_t vtbase = (size_t)(bb * H + h) * D * N;

  __shared__ unsigned short pbuf[4][2][16][32];

  // Q fragments: 2 q-tiles x 2 d-chunks
  H16 qa[2][2];
  #pragma unroll
  for (int qt = 0; qt < 2; qt++)
    #pragma unroll
    for (int c = 0; c < 2; c++) {
      const u16v8* p = (const u16v8*)(qkv + base + (size_t)(q0 + qt * 16 + ln) * C3 + c * 32 + 16 * half);
      qa[qt][c].h[0] = p[0];
      qa[qt][c].h[1] = p[1];
    }

  v8f ov[2][4];
  float ls[2][8];
  #pragma unroll
  for (int qt = 0; qt < 2; qt++) {
    #pragma unroll
    for (int dt = 0; dt < 4; dt++) ov[qt][dt] = vzero();
    #pragma unroll
    for (int r = 0; r < 8; r++) ls[qt][r] = 0.f;
  }

  for (int j0 = 0; j0 < N; j0 += 32) {
    // K fragments for two 16-key tiles x 2 d-chunks (contiguous per lane)
    H16 kb[2][2];
    #pragma unroll
    for (int t = 0; t < 2; t++)
      #pragma unroll
      for (int c = 0; c < 2; c++) {
        const u16v8* p = (const u16v8*)(qkv + base + 1024 + (size_t)(j0 + t * 16 + ln) * C3 + c * 32 + 16 * half);
        kb[t][c].h[0] = p[0];
        kb[t][c].h[1] = p[1];
      }
    #pragma unroll
    for (int qt = 0; qt < 2; qt++) {
      v8f s0 = vzero(), s1 = vzero();
      s0 = __builtin_amdgcn_wmma_f32_16x16x32_f16(false, qa[qt][0].v, false, kb[0][0].v, (short)0, s0, false, false);
      s0 = __builtin_amdgcn_wmma_f32_16x16x32_f16(false, qa[qt][1].v, false, kb[0][1].v, (short)0, s0, false, false);
      s1 = __builtin_amdgcn_wmma_f32_16x16x32_f16(false, qa[qt][0].v, false, kb[1][0].v, (short)0, s1, false, false);
      s1 = __builtin_amdgcn_wmma_f32_16x16x32_f16(false, qa[qt][1].v, false, kb[1][1].v, (short)0, s1, false, false);
      #pragma unroll
      for (int r = 0; r < 8; r++) {
        const float p0 = __expf(s0[r] * scale);
        const float p1 = __expf(s1[r] * scale);
        pbuf[wave][qt][r + 8 * half][ln]      = f2h_bits(p0);
        pbuf[wave][qt][r + 8 * half][16 + ln] = f2h_bits(p1);
        ls[qt][r] += p0 + p1;
      }
    }

    // per-wave LDS ordering: cross-lane read of this chunk's P
    asm volatile("s_wait_dscnt 0x0" ::: "memory");

    H16 pa[2];
    #pragma unroll
    for (int qt = 0; qt < 2; qt++)
      #pragma unroll
      for (int e = 0; e < 16; e++) {
        const int k = e + 16 * half;
        pa[qt].u[e] = pbuf[wave][qt][ln][k];
      }

    #pragma unroll
    for (int dt = 0; dt < 4; dt++) {
      H16 vb;   // b[e] = V[key][d] = vt[d][key], key = j0+16*half+e contiguous
      const u16v8* p = (const u16v8*)(vt + vtbase + (size_t)(dt * 16 + ln) * N + j0 + 16 * half);
      vb.h[0] = p[0];
      vb.h[1] = p[1];
      #pragma unroll
      for (int qt = 0; qt < 2; qt++)
        ov[qt][dt] = __builtin_amdgcn_wmma_f32_16x16x32_f16(false, pa[qt].v, false, vb.v, (short)0, ov[qt][dt], false, false);
    }
  }

  // one final cross-lane sum of the softmax denominators
  #pragma unroll
  for (int msk = 1; msk < 16; msk <<= 1)
    #pragma unroll
    for (int qt = 0; qt < 2; qt++)
      #pragma unroll
      for (int r = 0; r < 8; r++) ls[qt][r] += __shfl_xor(ls[qt][r], msk, 32);

  #pragma unroll
  for (int qt = 0; qt < 2; qt++)
    #pragma unroll
    for (int dt = 0; dt < 4; dt++)
      #pragma unroll
      for (int r = 0; r < 8; r++) {
        const float val = ov[qt][dt][r] / ls[qt][r];
        const int row = q0 + qt * 16 + r + 8 * half;
        o[(size_t)bb * N * C + (size_t)row * C + h * D + dt * 16 + ln] = f2h_bits(val);
      }
}

extern "C" void kernel_launch(void* const* d_in, const int* in_sizes, int n_in,
                              void* d_out, int out_size, void* d_ws, size_t ws_size,
                              hipStream_t stream)
{
  const int Bt = 4, Nt = 1024, Ct = 1024, Ht = 16, Lt = 6, FFt = 4096;
  const int M = Bt * Nt;                 // 4096 rows
  const float scale = 1.0f / 32.0f;      // C^-0.5, per reference (width**-0.5)

  const float* x    = (const float*)d_in[0];
  const float* Wqkv = (const float*)d_in[1];
  const float* Wout = (const float*)d_in[2];
  const float* bout = (const float*)d_in[3];
  const float* W1   = (const float*)d_in[4];
  const float* b1   = (const float*)d_in[5];
  const float* W2   = (const float*)d_in[6];
  const float* b2   = (const float*)d_in[7];
  const float* ln1g = (const float*)d_in[8];
  const float* ln1b = (const float*)d_in[9];
  const float* ln2g = (const float*)d_in[10];
  const float* ln2b = (const float*)d_in[11];

  // workspace layout (120 MB total)
  char* ws = (char*)d_ws;
  const size_t MB = (size_t)1024 * 1024;
  float*          x_cur  = (float*)(ws);                          // 16 MB f32 [M,C]
  unsigned short* h16    = (unsigned short*)(ws +  16 * MB);      //  8 MB f16 [M,C]
  unsigned short* qkv16  = (unsigned short*)(ws +  24 * MB);      // 24 MB f16 [M,3C]
  unsigned short* o16    = (unsigned short*)(ws +  48 * MB);      //  8 MB f16 [M,C]
  unsigned short* f16b   = (unsigned short*)(ws +  56 * MB);      // 32 MB f16 [M,FF]
  unsigned short* wtqkv  = (unsigned short*)(ws +  88 * MB);      //  6 MB f16 [3C,C]
  unsigned short* wtout  = (unsigned short*)(ws +  94 * MB);      //  2 MB f16 [C,C]
  unsigned short* wt1    = (unsigned short*)(ws +  96 * MB);      //  8 MB f16 [FF,C]
  unsigned short* wt2    = (unsigned short*)(ws + 104 * MB);      //  8 MB f16 [C,FF]
  unsigned short* vtbuf  = (unsigned short*)(ws + 112 * MB);      //  8 MB f16 [B*H,D,N]

  hipMemcpyAsync(x_cur, x, (size_t)M * Ct * sizeof(float), hipMemcpyDeviceToDevice, stream);

  for (int l = 0; l < Lt; l++) {
    // per-layer weight transpose+convert (tiny vs GEMM cost)
    wtrans_kernel<<<dim3(Ct / 64, (3 * Ct) / 64), 256, 0, stream>>>(
        Wqkv + (size_t)l * Ct * 3 * Ct, wtqkv, Ct, 3 * Ct);
    wtrans_kernel<<<dim3(Ct / 64, Ct / 64), 256, 0, stream>>>(
        Wout + (size_t)l * Ct * Ct, wtout, Ct, Ct);
    wtrans_kernel<<<dim3(Ct / 64, FFt / 64), 256, 0, stream>>>(
        W1 + (size_t)l * Ct * FFt, wt1, Ct, FFt);
    wtrans_kernel<<<dim3(FFt / 64, Ct / 64), 256, 0, stream>>>(
        W2 + (size_t)l * FFt * Ct, wt2, FFt, Ct);

    // attention
    ln_kernel<<<M, 256, 0, stream>>>(x_cur, ln1g, ln1b, h16, Ct);
    gemm_f16<0, 1><<<dim3(M / 128, (3 * Ct) / 128), 256, 0, stream>>>(
        h16, wtqkv, nullptr, nullptr, qkv16, M, Ct, 3 * Ct);
    vtrans_kernel<<<dim3(Bt * Ht, Nt / 64), 256, 0, stream>>>(qkv16, vtbuf);
    attn_kernel<<<Bt * Ht * (Nt / 128), 128, 0, stream>>>(qkv16, vtbuf, o16, scale);
    gemm_f16<0, 0><<<dim3(M / 128, Ct / 128), 256, 0, stream>>>(
        o16, wtout, bout + (size_t)l * Ct, x_cur, x_cur, M, Ct, Ct);

    // FFN
    ln_kernel<<<M, 256, 0, stream>>>(x_cur, ln2g, ln2b, h16, Ct);
    gemm_f16<1, 1><<<dim3(M / 128, FFt / 128), 256, 0, stream>>>(
        h16, wt1, b1 + (size_t)l * FFt, nullptr, f16b, M, Ct, FFt);
    float* outp = (l == Lt - 1) ? (float*)d_out : x_cur;
    gemm_f16<0, 0><<<dim3(M / 128, Ct / 128), 256, 0, stream>>>(
        f16b, wt2, b2 + (size_t)l * Ct, x_cur, outp, M, FFt, Ct);
  }
  (void)in_sizes; (void)n_in; (void)out_size; (void)ws_size;
}
